// MultiHeadAttention_34797825032534
// MI455X (gfx1250) — matrix operand order repaired and checked by reading, not compiled, not run on hardware
//
#include <hip/hip_runtime.h>

typedef _Float16 half_t;
typedef __attribute__((ext_vector_type(16))) _Float16 v16h;
typedef __attribute__((ext_vector_type(8)))  _Float16 v8h;
typedef __attribute__((ext_vector_type(4)))  _Float16 v4h;
typedef __attribute__((ext_vector_type(4)))  float    v4f;
typedef __attribute__((ext_vector_type(8)))  float    v8f;

// CDNA5 async global->LDS staging (ASYNCcnt-tracked), with compile-time
// feature detection so either toolchain lane still builds.
#if defined(__gfx1250__) &&                                             \
    __has_builtin(__builtin_amdgcn_global_load_async_to_lds_b128) &&     \
    __has_builtin(__builtin_amdgcn_s_wait_asynccnt)
#define USE_ASYNC_LDS 1
#else
#define USE_ASYNC_LDS 0
#endif

template <typename T> struct V4;
template <> struct V4<float>  { using type = v4f; };
template <> struct V4<half_t> { using type = v4h; };

static __device__ inline v8f zero_v8f() {
  v8f z;
#pragma unroll
  for (int i = 0; i < 8; ++i) z[i] = 0.0f;
  return z;
}

static __device__ inline v16h make_frag(v8h lo, v8h hi) {
  v16h r;
#pragma unroll
  for (int i = 0; i < 8; ++i) { r[i] = lo[i]; r[i + 8] = hi[i]; }
  return r;
}

// A-fragment (16x32 f16, MxK): lane holds row M = lane&15.
// K per half h: K = 2*koff + h (contiguous run), koff = 8*(lane>=16).
static __device__ inline v16h load_a_frag(const half_t* rowPtr, int koff) {
  v8h lo = *(const v8h*)(rowPtr + 2 * koff);
  v8h hi = *(const v8h*)(rowPtr + 2 * koff + 8);
  return make_frag(lo, hi);
}

// B-fragment (32x16 f16, KxN) from K-major (transposed) storage:
// lane holds col N = lane&15; K = 16*(lane>=16) + h  => one contiguous run.
static __device__ inline v16h load_b_frag(const half_t* p) {
  v8h lo = *(const v8h*)(p);
  v8h hi = *(const v8h*)(p + 8);
  return make_frag(lo, hi);
}

#if USE_ASYNC_LDS
// Builtin signature (from hipcc diagnostic): param 0 is
// 'int __attribute__((vector_size(16)))' pointer in the global AS.
typedef int async_v4i __attribute__((vector_size(16)));
typedef __attribute__((address_space(1))) async_v4i* async_gptr;
typedef __attribute__((address_space(3))) async_v4i* async_lptr;
static __device__ inline void async_copy_b128(const half_t* gsrc, half_t* ldst) {
  __builtin_amdgcn_global_load_async_to_lds_b128(
      (async_gptr)const_cast<half_t*>(gsrc), (async_lptr)ldst, 0, 0);
}
#endif

// ---------------------------------------------------------------------------
// C = A[MxK] * B[KxN], f16 WMMA / f32 accumulate.
// Block tile 128x128, BK=32, 256 threads = 8 waves (4Mx2N), wave tile 32x64.
// Double-buffered LDS; next tile prefetched into registers during compute;
// tile t+2 prefetched into GL2 via global_prefetch_b8.
// ---------------------------------------------------------------------------
template <typename AT, typename BT, typename OutT>
__global__ __launch_bounds__(256) void gemm_wmma_kernel(
    const AT* __restrict__ A, const BT* __restrict__ Bm, OutT* __restrict__ C,
    int M, int N, int K) {
  __shared__ half_t Alds[2][128][40];     // row-major A tile  (80B rows, 16B aligned)
  __shared__ half_t BldsT[2][128][40];    // transposed B tile: [n][k]

  const int tid   = threadIdx.x;
  const int lane  = tid & 31;
  const int wid   = tid >> 5;
  const int waveM = wid >> 1;             // 0..3 -> 32-row stripe
  const int waveN = wid & 1;              // 0..1 -> 64-col stripe
  const int lo    = lane & 15;
  const int hi    = lane >> 4;
  const int m0    = blockIdx.y * 128;
  const int n0    = blockIdx.x * 128;
  const int koffA = hi ? 8 : 0;
  const int kgB   = hi ? 16 : 0;

  using AV4 = typename V4<AT>::type;
  using BV4 = typename V4<BT>::type;

  v8f acc[2][4];
#pragma unroll
  for (int mi = 0; mi < 2; ++mi)
#pragma unroll
    for (int ni = 0; ni < 4; ++ni) acc[mi][ni] = zero_v8f();

  const int nk = K / 32;

  // Prologue: stage tile 0 into buffer 0 (load + f16 convert + deposit).
  {
#pragma unroll
    for (int j = 0; j < 4; ++j) {               // A: 128x32 = 1024 4-elem chunks
      int chunk = tid + j * 256;
      int r  = chunk >> 3;
      int c4 = (chunk & 7) * 4;
      AV4 x = *(const AV4*)(A + (size_t)(m0 + r) * K + c4);
      half_t* p = &Alds[0][r][c4];
      p[0] = (half_t)x[0]; p[1] = (half_t)x[1];
      p[2] = (half_t)x[2]; p[3] = (half_t)x[3];
    }
#pragma unroll
    for (int j = 0; j < 4; ++j) {               // B: 32x128, scatter-transpose
      int chunk = tid + j * 256;
      int kr = chunk >> 5;
      int c4 = (chunk & 31) * 4;
      BV4 x = *(const BV4*)(Bm + (size_t)kr * N + n0 + c4);
      BldsT[0][c4 + 0][kr] = (half_t)x[0];
      BldsT[0][c4 + 1][kr] = (half_t)x[1];
      BldsT[0][c4 + 2][kr] = (half_t)x[2];
      BldsT[0][c4 + 3][kr] = (half_t)x[3];
    }
  }
  __syncthreads();

  for (int t = 0; t < nk; ++t) {
    const int buf = t & 1;
    const bool hasNext = (t + 1 < nk);

    // Prefetch next tile into registers (VMEM overlaps matrix pipe below).
    AV4 aReg[4];
    BV4 bReg[4];
    if (hasNext) {
      const int kt = (t + 1) * 32;
#pragma unroll
      for (int j = 0; j < 4; ++j) {
        int chunk = tid + j * 256;
        int r  = chunk >> 3;
        int c4 = (chunk & 7) * 4;
        aReg[j] = *(const AV4*)(A + (size_t)(m0 + r) * K + kt + c4);
      }
#pragma unroll
      for (int j = 0; j < 4; ++j) {
        int chunk = tid + j * 256;
        int kr = chunk >> 5;
        int c4 = (chunk & 31) * 4;
        bReg[j] = *(const BV4*)(Bm + (size_t)(kt + kr) * N + n0 + c4);
      }
      // L2 prefetch two tiles ahead (global_prefetch_b8).
      const int kt2 = kt + 32;
      if (kt2 < K) {
        __builtin_prefetch(A + (size_t)(m0 + (tid >> 1)) * K + kt2, 0, 1);
        __builtin_prefetch(Bm + (size_t)(kt2 + (tid >> 3)) * N + n0 + (tid & 7) * 16, 0, 1);
      }
    }

    // Compute on current buffer: 8 WMMAs per wave per k-step.
    v16h afrag[2], bfrag[4];
#pragma unroll
    for (int mi = 0; mi < 2; ++mi)
      afrag[mi] = load_a_frag(&Alds[buf][waveM * 32 + mi * 16 + lo][0], koffA);
#pragma unroll
    for (int ni = 0; ni < 4; ++ni)
      bfrag[ni] = load_b_frag(&BldsT[buf][waveN * 64 + ni * 16 + lo][kgB]);
#pragma unroll
    for (int mi = 0; mi < 2; ++mi)
#pragma unroll
      for (int ni = 0; ni < 4; ++ni)
        acc[mi][ni] = __builtin_amdgcn_wmma_f32_16x16x32_f16(
            false, afrag[mi], false, bfrag[ni], (short)0, acc[mi][ni],
            false, false);

    if (hasNext) {
      __syncthreads();   // all waves done reading buf^1 (finished last iter)
      const int nbuf = buf ^ 1;
#pragma unroll
      for (int j = 0; j < 4; ++j) {
        int chunk = tid + j * 256;
        int r  = chunk >> 3;
        int c4 = (chunk & 7) * 4;
        half_t* p = &Alds[nbuf][r][c4];
        p[0] = (half_t)aReg[j][0]; p[1] = (half_t)aReg[j][1];
        p[2] = (half_t)aReg[j][2]; p[3] = (half_t)aReg[j][3];
      }
#pragma unroll
      for (int j = 0; j < 4; ++j) {
        int chunk = tid + j * 256;
        int kr = chunk >> 5;
        int c4 = (chunk & 31) * 4;
        BldsT[nbuf][c4 + 0][kr] = (half_t)bReg[j][0];
        BldsT[nbuf][c4 + 1][kr] = (half_t)bReg[j][1];
        BldsT[nbuf][c4 + 2][kr] = (half_t)bReg[j][2];
        BldsT[nbuf][c4 + 3][kr] = (half_t)bReg[j][3];
      }
      __syncthreads();
    }
  }

  // C layout: lane holds col N = lane&15, rows M = v + 8*(lane>=16).
#pragma unroll
  for (int mi = 0; mi < 2; ++mi)
#pragma unroll
    for (int ni = 0; ni < 4; ++ni)
#pragma unroll
      for (int vv = 0; vv < 8; ++vv) {
        int r = m0 + waveM * 32 + mi * 16 + vv + 8 * hi;
        int c = n0 + waveN * 64 + ni * 16 + lo;
        C[(size_t)r * N + c] = (OutT)acc[mi][ni][vv];
      }
}

#define S_LEN  2048
#define DMODEL 2048
#define DH     128

// ---------------------------------------------------------------------------
// Flash attention: block = one (b,h) + 64 query rows; 4 waves x 16 rows.
// Q/K tiles staged via async global->LDS (ASYNCcnt); V via registers
// (needs f16 scatter-transpose).
// ---------------------------------------------------------------------------
__global__ __launch_bounds__(128) void flash_attn_kernel(
    const half_t* __restrict__ Xq, const half_t* __restrict__ Xk,
    const half_t* __restrict__ Xv, half_t* __restrict__ Out) {
  __shared__ half_t Qlds[64][136];    // 64 q-rows x 128 dims
  __shared__ half_t Klds[64][136];    // row-major K tile == B^T for Q.K^T
  __shared__ half_t VldsT[128][72];   // transposed V: [d][key]
  __shared__ half_t Plds[64][72];     // per-wave P staging (C-layout -> A-layout)

  const int tid  = threadIdx.x;       // 0..127
  const int lane = tid & 31;
  const int w    = tid >> 5;          // wave 0..3
  const int lo   = lane & 15;
  const int hi   = lane >> 4;
  const int qblk = blockIdx.x;
  const int h    = blockIdx.y;
  const int b    = blockIdx.z;
  const int q0   = qblk * 64;
  const size_t baseOff = (size_t)b * S_LEN * DMODEL + (size_t)h * DH;

  // Stage Q tile (64x128): 1024 8-half chunks / 128 threads.
#pragma unroll
  for (int j = 0; j < 8; ++j) {
    int chunk = tid + j * 128;
    int r  = chunk >> 4;
    int c8 = (chunk & 15) * 8;
    const half_t* src = Xq + baseOff + (size_t)(q0 + r) * DMODEL + c8;
#if USE_ASYNC_LDS
    async_copy_b128(src, &Qlds[r][c8]);
#else
    *(v8h*)&Qlds[r][c8] = *(const v8h*)src;
#endif
  }
#if USE_ASYNC_LDS
  __builtin_amdgcn_s_wait_asynccnt(0);
#endif
  __syncthreads();

  const int koffA = hi ? 8 : 0;
  const int kgB   = hi ? 16 : 0;
  const int qrow0 = q0 + w * 16;

  v16h qfrag[4];
#pragma unroll
  for (int ks = 0; ks < 4; ++ks)
    qfrag[ks] = load_a_frag(&Qlds[w * 16 + lo][ks * 32], koffA);

  v8f Oacc[8];
#pragma unroll
  for (int d = 0; d < 8; ++d) Oacc[d] = zero_v8f();
  float run_m[8], run_l[8];
#pragma unroll
  for (int vv = 0; vv < 8; ++vv) { run_m[vv] = -1e30f; run_l[vv] = 0.0f; }

  for (int t = 0; t <= qblk; ++t) {   // causal: only tiles with kb <= q0
    const int kb = t * 64;
    __syncthreads();                  // protect Klds/VldsT from prior readers
#pragma unroll
    for (int j = 0; j < 8; ++j) {
      int chunk = tid + j * 128;
      int r  = chunk >> 4;
      int c8 = (chunk & 15) * 8;
      const half_t* ksrc = Xk + baseOff + (size_t)(kb + r) * DMODEL + c8;
#if USE_ASYNC_LDS
      async_copy_b128(ksrc, &Klds[r][c8]);   // K: straight copy, ASYNCcnt path
#else
      *(v8h*)&Klds[r][c8] = *(const v8h*)ksrc;
#endif
      v8h vv8 = *(const v8h*)(Xv + baseOff + (size_t)(kb + r) * DMODEL + c8);
#pragma unroll
      for (int e = 0; e < 8; ++e) VldsT[c8 + e][r] = vv8[e];
    }
#if USE_ASYNC_LDS
    __builtin_amdgcn_s_wait_asynccnt(0);
#endif
    __syncthreads();

    // S = Q K^T: 4 key subtiles x (Dh=128 in 4 WMMA steps).
    v8f s[4];
#pragma unroll
    for (int ns = 0; ns < 4; ++ns) {
      v8f a = zero_v8f();
#pragma unroll
      for (int ks = 0; ks < 4; ++ks) {
        v16h bfr = load_b_frag(&Klds[ns * 16 + lo][ks * 32 + kgB]);
        a = __builtin_amdgcn_wmma_f32_16x16x32_f16(
            false, qfrag[ks], false, bfr, (short)0, a, false, false);
      }
      s[ns] = a;
    }
    // Scale + causal mask.
    const float scale = 0.0883883476483184f;  // 1/sqrt(128)
#pragma unroll
    for (int ns = 0; ns < 4; ++ns)
#pragma unroll
      for (int vv = 0; vv < 8; ++vv) {
        int row = qrow0 + vv + 8 * hi;
        int col = kb + ns * 16 + lo;
        float x = s[ns][vv] * scale;
        s[ns][vv] = (col > row) ? -1e30f : x;
      }
    // Online softmax: rows live across 16-lane half-groups; shfl_xor reduce.
#pragma unroll
    for (int vv = 0; vv < 8; ++vv) {
      float mx = fmaxf(fmaxf(s[0][vv], s[1][vv]), fmaxf(s[2][vv], s[3][vv]));
#pragma unroll
      for (int msk = 1; msk < 16; msk <<= 1) mx = fmaxf(mx, __shfl_xor(mx, msk, 32));
      float m_new = fmaxf(run_m[vv], mx);
      float corr  = __expf(run_m[vv] - m_new);
      run_m[vv] = m_new;
      float lsum = 0.0f;
#pragma unroll
      for (int ns = 0; ns < 4; ++ns) {
        float p = __expf(s[ns][vv] - m_new);
        s[ns][vv] = p;
        lsum += p;
      }
#pragma unroll
      for (int msk = 1; msk < 16; msk <<= 1) lsum += __shfl_xor(lsum, msk, 32);
      run_l[vv] = run_l[vv] * corr + lsum;
#pragma unroll
      for (int d = 0; d < 8; ++d) Oacc[d][vv] *= corr;
    }
    // P (C-layout) -> LDS -> A-layout. Same-wave DS ops are in-order and each
    // wave touches only its own Plds rows: no barrier needed.
#pragma unroll
    for (int ns = 0; ns < 4; ++ns)
#pragma unroll
      for (int vv = 0; vv < 8; ++vv)
        Plds[w * 16 + vv + 8 * hi][ns * 16 + lo] = (half_t)s[ns][vv];
    // O += P V: 64 keys in 2 WMMA k-steps, 8 d-subtiles.
#pragma unroll
    for (int ks2 = 0; ks2 < 2; ++ks2) {
      v16h pfrag = load_a_frag(&Plds[w * 16 + lo][ks2 * 32], koffA);
#pragma unroll
      for (int d = 0; d < 8; ++d) {
        v16h vfr = load_b_frag(&VldsT[d * 16 + lo][ks2 * 32 + kgB]);
        Oacc[d] = __builtin_amdgcn_wmma_f32_16x16x32_f16(
            false, pfrag, false, vfr, (short)0, Oacc[d], false, false);
      }
    }
  }

  // Normalize and store f16 output at [b, row, h*128 + d].
#pragma unroll
  for (int vv = 0; vv < 8; ++vv) {
    float inv = 1.0f / run_l[vv];
    int row = qrow0 + vv + 8 * hi;
#pragma unroll
    for (int d = 0; d < 8; ++d)
      Out[baseOff + (size_t)row * DMODEL + d * 16 + lo] = (half_t)(Oacc[d][vv] * inv);
  }
}

extern "C" void kernel_launch(void* const* d_in, const int* in_sizes, int n_in,
                              void* d_out, int out_size, void* d_ws, size_t ws_size,
                              hipStream_t stream) {
  (void)in_sizes; (void)n_in; (void)out_size; (void)ws_size;
  const int B = 2, S = 2048, D = 2048, H = 16;
  const float* q  = (const float*)d_in[0];
  const float* k  = (const float*)d_in[1];
  const float* v  = (const float*)d_in[2];
  const float* wq = (const float*)d_in[3];
  const float* wk = (const float*)d_in[4];
  const float* wv = (const float*)d_in[5];
  const float* wo = (const float*)d_in[6];

  const size_t BSD = (size_t)B * S * D;          // 8,388,608 elements
  half_t* xq16   = (half_t*)d_ws;                // 4 x BSD f16 = 67 MB total
  half_t* xk16   = xq16 + BSD;
  half_t* xv16   = xk16 + BSD;
  half_t* attn16 = xv16 + BSD;

  dim3 gblk(256);
  dim3 ggrid(D / 128, (B * S) / 128);            // 16 x 32 blocks

  // Projections: f32 in (f16 conversion fused into LDS staging), f16 out.
  gemm_wmma_kernel<float, float, half_t><<<ggrid, gblk, 0, stream>>>(q, wq, xq16, B * S, D, D);
  gemm_wmma_kernel<float, float, half_t><<<ggrid, gblk, 0, stream>>>(k, wk, xk16, B * S, D, D);
  gemm_wmma_kernel<float, float, half_t><<<ggrid, gblk, 0, stream>>>(v, wv, xv16, B * S, D, D);

  // Causal flash attention, all-WMMA.
  dim3 agrid(S / 64, H, B);
  flash_attn_kernel<<<agrid, dim3(128), 0, stream>>>(xq16, xk16, xv16, attn16);

  // Output projection -> f32 d_out.
  gemm_wmma_kernel<half_t, float, float><<<ggrid, gblk, 0, stream>>>(attn16, wo, (float*)d_out, B * S, D, D);
}
